// VectorQuantizer_88931592831505
// MI455X (gfx1250) — matrix operand order repaired and checked by reading, not compiled, not run on hardware
//
#include <hip/hip_runtime.h>
#include <hip/hip_bf16.h>
#include <stdint.h>

typedef __bf16 bf16;
typedef bf16  v16bf __attribute__((ext_vector_type(16)));
typedef float v8f   __attribute__((ext_vector_type(8)));
typedef unsigned int v4u __attribute__((ext_vector_type(4)));
typedef int  v8i __attribute__((ext_vector_type(8)));
typedef int  v4i __attribute__((ext_vector_type(4)));

#define N_ROWS  32768
#define K_CODES 8192
#define D_DIM   512

// ---------- helpers ----------

__device__ __forceinline__ bf16 f2bf(float f) {
  unsigned u = __builtin_bit_cast(unsigned, f);
  unsigned r = (u + 0x7FFFu + ((u >> 16) & 1u)) >> 16;   // round-to-nearest-even
  unsigned short h = (unsigned short)r;
  return __builtin_bit_cast(bf16, h);
}

__device__ __forceinline__ unsigned lds_lo32(const void* p) {
  // low 32 bits of a generic pointer into LDS == LDS byte address (ISA 10.2)
  return (unsigned)(unsigned long long)(uintptr_t)p;
}

// Issue a 2D TDM tile load Global->LDS. D# built per CDNA5 ISA
// group0/group1 bitfields; groups 2/3 zero (2D tensor). data_size=2 bytes.
// This toolchain's builtin takes 6 args: (v4u g0, v8i g1, v4i g2, v4i g3, v8i extra, i32 cpol)
__device__ __forceinline__ void tdm_load_2d(unsigned lds_byte_addr, const void* gptr,
                                            unsigned tensor_d0, unsigned tensor_d1,
                                            unsigned tile_d0, unsigned tile_d1,
                                            unsigned stride0_elems) {
  unsigned long long ga = (unsigned long long)(uintptr_t)gptr;
  v4u g0;
  g0[0] = 1u;                                              // count=1, user D#
  g0[1] = lds_byte_addr;                                   // lds_addr
  g0[2] = (unsigned)(ga & 0xFFFFFFFFull);                  // global_addr[31:0]
  g0[3] = (unsigned)((ga >> 32) & 0x01FFFFFFull) | (2u << 30); // addr[56:32] | type=2
  v8i g1;
  g1[0] = (int)(1u << 16);                                 // data_size=1 (2 bytes)
  g1[1] = (int)((tensor_d0 & 0xFFFFu) << 16);              // tensor_dim0 lo
  g1[2] = (int)((tensor_d0 >> 16) | ((tensor_d1 & 0xFFFFu) << 16));
  g1[3] = (int)((tensor_d1 >> 16) | (tile_d0 << 16));      // tile_dim0
  g1[4] = (int)(tile_d1 & 0xFFFFu);                        // tile_dim1 (tile_dim2=0)
  g1[5] = (int)stride0_elems;                              // tensor_dim0_stride lo
  g1[6] = 0;
  g1[7] = 0;                                               // dim1 stride unused (2D)
  v4i z4; z4[0] = 0; z4[1] = 0; z4[2] = 0; z4[3] = 0;
  v8i z8; z8[0] = 0; z8[1] = 0; z8[2] = 0; z8[3] = 0;
  z8[4] = 0; z8[5] = 0; z8[6] = 0; z8[7] = 0;
  __builtin_amdgcn_tensor_load_to_lds(g0, g1, z4, z4, z8, 0);
}

// ---------- phase P: fp32 -> bf16 convert + row sum-of-squares ----------
__global__ __launch_bounds__(256) void vq_prep_kernel(
    const float* __restrict__ src, bf16* __restrict__ dst,
    float* __restrict__ sumsq, int rows)
{
  const int wave = threadIdx.x >> 5;
  const int lane = threadIdx.x & 31;
  const int row  = blockIdx.x * 8 + wave;
  if (row >= rows) return;
  const float* s = src + (size_t)row * D_DIM;
  bf16*        d = dst + (size_t)row * D_DIM;
  float acc = 0.f;
  #pragma unroll
  for (int i = 0; i < D_DIM / 32; ++i) {
    int c = lane + i * 32;
    float f = s[c];
    d[c] = f2bf(f);
    acc = fmaf(f, f, acc);
  }
  #pragma unroll
  for (int mask = 1; mask <= 16; mask <<= 1) acc += __shfl_xor(acc, mask, 32);
  if (lane == 0) sumsq[row] = acc;
}

__global__ void vq_zero_kernel(float* p, int n) {
  int i = blockIdx.x * blockDim.x + threadIdx.x;
  if (i < n) p[i] = 0.f;
}

// ---------- fused distance GEMM, double-buffered TDM pipeline ----------
// PHASE 0: online softmax stats (m,s,t) + argmin per row -> mrow/srow/idxr/out_idx/entp
// PHASE 1: recompute logits, accumulate avg_probs[k] += exp(l-m)/s   (atomic)
template<int PHASE>
__global__ __launch_bounds__(256) void vq_gemm_kernel(
    const bf16* __restrict__ xbf, const bf16* __restrict__ cbf,
    const float* __restrict__ a2, const float* __restrict__ b2,
    float* __restrict__ mrow, float* __restrict__ srow,
    int* __restrict__ idxr, float* __restrict__ out_idx_f,
    float* __restrict__ avgp, float* __restrict__ entp)
{
  __shared__ bf16 xs[32 * D_DIM];     // x block tile (bf16), staged once (32 KB)
  __shared__ bf16 cs[2][64 * 64];     // codebook tile, double buffered (2 x 8 KB)
  __shared__ float sm[4][32], ssv[4][32], stv[4][32], sdv[4][32];
  __shared__ int   siv[4][32];

  const int tid  = threadIdx.x;
  const int lane = tid & 31;
  const int wave = tid >> 5;
  const int wr   = wave >> 2;         // 0..1 : row sub-tile
  const int wc   = wave & 3;          // 0..3 : col sub-tile
  const int hf   = lane >> 4;
  const int l16  = lane & 15;
  const int row0 = blockIdx.x * 32;

  // stage x tile + first codebook tile; overlap nothing yet (prologue)
  if (tid < 32) {
    tdm_load_2d(lds_lo32(xs), xbf + (size_t)row0 * D_DIM,
                D_DIM, N_ROWS, D_DIM, 32, D_DIM);
    tdm_load_2d(lds_lo32(cs[0]), cbf,                 // kt=0, dt=0
                D_DIM, K_CODES, 64, 64, D_DIM);
  }
  __builtin_amdgcn_s_wait_tensorcnt(0);
  __syncthreads();

  int rloc[8]; float a2r[8];
  float m_st[8], s_st[8], t_st[8], dmin[8]; int imin[8];
  float msv[8], rsv[8];
  #pragma unroll
  for (int v = 0; v < 8; ++v) {
    rloc[v] = wr * 16 + hf * 8 + v;             // C-matrix row = v + 8*half
    a2r[v]  = a2[row0 + rloc[v]];
    m_st[v] = -1e30f; s_st[v] = 0.f; t_st[v] = 0.f;
    dmin[v] = 3.0e38f; imin[v] = 0;
    if constexpr (PHASE == 1) {
      msv[v] = mrow[row0 + rloc[v]];
      rsv[v] = 1.0f / srow[row0 + rloc[v]];
    }
  }

  for (int kt = 0; kt < K_CODES / 64; ++kt) {
    v8f c = {};
    const int colabs = kt * 64 + wc * 16 + l16;  // C-matrix col = lane%16
    float b2c = b2[colabs];
    #pragma unroll
    for (int dt = 0; dt < D_DIM / 64; ++dt) {
      // ---- prefetch next tile into the other buffer (overlaps WMMA below) ----
      // buffer parity of tile t = kt*8+dt is dt&1 (kt*8 is even)
      const bool last = (dt == 7) && (kt == K_CODES / 64 - 1);
      if (!last && tid < 32) {
        const int nkt = (dt == 7) ? kt + 1 : kt;
        const int ndt = (dt + 1) & 7;
        tdm_load_2d(lds_lo32(cs[(dt + 1) & 1]),
                    cbf + (size_t)(nkt * 64) * D_DIM + ndt * 64,
                    D_DIM, K_CODES, 64, 64, D_DIM);
      }
      // ---- compute on current buffer ----
      const bf16* csb = cs[dt & 1];
      #pragma unroll
      for (int ds = 0; ds < 2; ++ds) {
        v16bf af, bfr;
        {   // A 16x32 bf16 layout (ISA 7.12.2)
          const bf16* p = xs + (wr * 16 + l16) * D_DIM + dt * 64 + ds * 32;
          #pragma unroll
          for (int j = 0; j < 8; ++j) {
            int k = ((j < 4) ? 2 * j : 2 * j + 8) + hf * 8;
            af[2 * j]     = p[k];
            af[2 * j + 1] = p[k + 1];
          }
        }
        {   // B 32x16 bf16: lane holds one column's 16 K-values
          const bf16* p = csb + (wc * 16 + l16) * 64 + ds * 32 + hf * 8;
          #pragma unroll
          for (int j = 0; j < 8; ++j) {
            bfr[2 * j]     = p[(j < 4) ? 2 * j : 2 * j + 8];
            bfr[2 * j + 1] = p[((j < 4) ? 2 * j : 2 * j + 8) + 1];
          }
        }
        c = __builtin_amdgcn_wmma_f32_16x16x32_bf16(false, af, false, bfr,
                                                    (short)0, c, false, false);
      }
      // next tile must have landed; one barrier closes the stage
      __builtin_amdgcn_s_wait_tensorcnt(0);
      __syncthreads();
    }
    float pc = 0.f;
    #pragma unroll
    for (int v = 0; v < 8; ++v) {
      float dist = a2r[v] - 2.0f * c[v] + b2c;
      float l = -dist;                           // logits, T = 1
      if constexpr (PHASE == 0) {
        if (dist < dmin[v]) { dmin[v] = dist; imin[v] = colabs; }
        if (l > m_st[v]) {
          float sc = __expf(m_st[v] - l);
          t_st[v] = sc * (t_st[v] + (m_st[v] - l) * s_st[v]);
          s_st[v] = sc * s_st[v] + 1.0f;
          m_st[v] = l;
        } else {
          float e = __expf(l - m_st[v]);
          s_st[v] += e;
          t_st[v] += e * (l - m_st[v]);
        }
      } else {
        pc += __expf(l - msv[v]) * rsv[v];
      }
    }
    if constexpr (PHASE == 1) {
      pc += __shfl_xor(pc, 16, 32);              // combine both row halves
      if (hf == 0) atomicAdd(&avgp[colabs], pc);
    }
  }

  if constexpr (PHASE == 0) {
    // reduce across the 16 lanes (columns) inside each half-wave
    #pragma unroll
    for (int mask = 1; mask <= 8; mask <<= 1) {
      #pragma unroll
      for (int v = 0; v < 8; ++v) {
        float m2 = __shfl_xor(m_st[v], mask, 32);
        float s2 = __shfl_xor(s_st[v], mask, 32);
        float t2 = __shfl_xor(t_st[v], mask, 32);
        float d2 = __shfl_xor(dmin[v], mask, 32);
        int   i2 = __shfl_xor(imin[v], mask, 32);
        float mm = fmaxf(m_st[v], m2);
        float e1 = __expf(m_st[v] - mm), e2 = __expf(m2 - mm);
        t_st[v] = e1 * (t_st[v] + (m_st[v] - mm) * s_st[v]) + e2 * (t2 + (m2 - mm) * s2);
        s_st[v] = e1 * s_st[v] + e2 * s2;
        m_st[v] = mm;
        if (d2 < dmin[v] || (d2 == dmin[v] && i2 < imin[v])) { dmin[v] = d2; imin[v] = i2; }
      }
    }
    if (l16 == 0) {
      #pragma unroll
      for (int v = 0; v < 8; ++v) {
        int r = rloc[v];
        sm[wc][r] = m_st[v]; ssv[wc][r] = s_st[v]; stv[wc][r] = t_st[v];
        sdv[wc][r] = dmin[v]; siv[wc][r] = imin[v];
      }
    }
    __syncthreads();
    if (tid < 32) {                               // wave 0 merges the 4 col-groups
      int r = tid;
      float M = sm[0][r], S = ssv[0][r], T = stv[0][r], Dd = sdv[0][r];
      int I = siv[0][r];
      #pragma unroll
      for (int w = 1; w < 4; ++w) {
        float m2 = sm[w][r], s2 = ssv[w][r], t2 = stv[w][r], d2 = sdv[w][r];
        int i2 = siv[w][r];
        float mm = fmaxf(M, m2);
        float e1 = __expf(M - mm), e2 = __expf(m2 - mm);
        T = e1 * (T + (M - mm) * S) + e2 * (t2 + (m2 - mm) * s2);
        S = e1 * S + e2 * s2;
        M = mm;
        if (d2 < Dd || (d2 == Dd && i2 < I)) { Dd = d2; I = i2; }
      }
      mrow[row0 + r] = M;
      srow[row0 + r] = S;
      idxr[row0 + r] = I;
      out_idx_f[row0 + r] = (float)I;
      float entc = T / S - __logf(S);             // sum_k p*log p for this row
      #pragma unroll
      for (int mask = 1; mask <= 16; mask <<= 1) entc += __shfl_xor(entc, mask, 32);
      if (tid == 0) entp[blockIdx.x] = entc;
    }
  }
}

// ---------- gather quantized rows + squared error partials ----------
__global__ __launch_bounds__(256) void vq_gather_kernel(
    const float* __restrict__ x, const float* __restrict__ cb,
    const int* __restrict__ idxr, float* __restrict__ outq,
    float* __restrict__ sqp)
{
  const int row = blockIdx.x;
  const int tid = threadIdx.x;
  const int idx = idxr[row];
  const float* c  = cb + (size_t)idx * D_DIM;
  const float* xr = x  + (size_t)row * D_DIM;
  float* o = outq + (size_t)row * D_DIM;
  float sq = 0.f;
  #pragma unroll
  for (int i = 0; i < D_DIM / 256; ++i) {
    int d = tid + i * 256;
    float q = c[d];
    o[d] = q;                                    // quantized_ste == quantized fwd
    float df = q - xr[d];
    sq = fmaf(df, df, sq);
  }
  #pragma unroll
  for (int mask = 1; mask <= 16; mask <<= 1) sq += __shfl_xor(sq, mask, 32);
  __shared__ float red[8];
  if ((tid & 31) == 0) red[tid >> 5] = sq;
  __syncthreads();
  if (tid == 0) {
    float s = 0.f;
    #pragma unroll
    for (int w = 0; w < 8; ++w) s += red[w];
    sqp[row] = s;
  }
}

// ---------- final scalar losses ----------
__global__ __launch_bounds__(256) void vq_final_kernel(
    const float* __restrict__ avgp, const float* __restrict__ entp,
    const float* __restrict__ sqp, float* __restrict__ out_scalars)
{
  const int tid = threadIdx.x;
  float aent = 0.f, sent = 0.f, sq = 0.f;
  for (int k = tid; k < K_CODES; k += 256) {
    float a = avgp[k] * (1.0f / (float)N_ROWS);
    aent += a * __logf(a + 1e-5f);
  }
  for (int i = tid; i < N_ROWS / 32; i += 256) sent += entp[i];
  for (int i = tid; i < N_ROWS; i += 256)      sq   += sqp[i];
  __shared__ float r1[256], r2[256], r3[256];
  r1[tid] = aent; r2[tid] = sent; r3[tid] = sq;
  __syncthreads();
  for (int off = 128; off > 0; off >>= 1) {
    if (tid < off) { r1[tid] += r1[tid+off]; r2[tid] += r2[tid+off]; r3[tid] += r3[tid+off]; }
    __syncthreads();
  }
  if (tid == 0) {
    float avg_entropy    = -r1[0];
    float sample_entropy = -(r2[0] / (float)N_ROWS);
    float entropy_loss   = 0.1f * (sample_entropy - avg_entropy);
    float msq = r3[0] / ((float)N_ROWS * (float)D_DIM);
    float e_l = 0.25f * msq;
    float q_l = msq;
    out_scalars[0] = e_l + q_l + entropy_loss;
    out_scalars[1] = e_l;
    out_scalars[2] = q_l;
    out_scalars[3] = entropy_loss;
  }
}

// ---------- launch ----------
extern "C" void kernel_launch(void* const* d_in, const int* in_sizes, int n_in,
                              void* d_out, int out_size, void* d_ws, size_t ws_size,
                              hipStream_t stream)
{
  const float* x  = (const float*)d_in[0];   // [32768, 512] fp32
  const float* cb = (const float*)d_in[1];   // [8192, 512] fp32

  char* ws = (char*)d_ws;
  size_t off = 0;
  auto alloc = [&](size_t bytes) {
    void* p = ws + off;
    off = (off + bytes + 255) & ~(size_t)255;
    return p;
  };
  bf16*  xbf  = (bf16*)alloc((size_t)N_ROWS  * D_DIM * sizeof(bf16));
  bf16*  cbf  = (bf16*)alloc((size_t)K_CODES * D_DIM * sizeof(bf16));
  float* a2   = (float*)alloc((size_t)N_ROWS * sizeof(float));
  float* b2   = (float*)alloc((size_t)K_CODES * sizeof(float));
  float* mrow = (float*)alloc((size_t)N_ROWS * sizeof(float));
  float* srow = (float*)alloc((size_t)N_ROWS * sizeof(float));
  int*   idxr = (int*)  alloc((size_t)N_ROWS * sizeof(int));
  float* avgp = (float*)alloc((size_t)K_CODES * sizeof(float));
  float* entp = (float*)alloc((size_t)(N_ROWS / 32) * sizeof(float));
  float* sqp  = (float*)alloc((size_t)N_ROWS * sizeof(float));

  float* outq        = (float*)d_out;                    // [N, D]
  float* out_idx_f   = outq + (size_t)N_ROWS * D_DIM;    // [N] indices as float
  float* out_scalars = out_idx_f + N_ROWS;               // loss, e, q, ent

  vq_prep_kernel<<<N_ROWS / 8, 256, 0, stream>>>(x,  xbf, a2, N_ROWS);
  vq_prep_kernel<<<K_CODES / 8, 256, 0, stream>>>(cb, cbf, b2, K_CODES);
  vq_zero_kernel<<<(K_CODES + 255) / 256, 256, 0, stream>>>(avgp, K_CODES);
  vq_gemm_kernel<0><<<N_ROWS / 32, 256, 0, stream>>>(xbf, cbf, a2, b2, mrow, srow,
                                                     idxr, out_idx_f, avgp, entp);
  vq_gemm_kernel<1><<<N_ROWS / 32, 256, 0, stream>>>(xbf, cbf, a2, b2, mrow, srow,
                                                     idxr, out_idx_f, avgp, entp);
  vq_gather_kernel<<<N_ROWS, 256, 0, stream>>>(x, cb, idxr, outq, sqp);
  vq_final_kernel<<<1, 256, 0, stream>>>(avgp, entp, sqp, out_scalars);
}
